// FlashSelfAttention_90477781057575
// MI455X (gfx1250) — matrix operand order repaired and checked by reading, not compiled, not run on hardware
//
#include <hip/hip_runtime.h>

#define D_MODEL  1024
#define N_HEADS  16
#define HEAD_DIM 64
#define SEQ      2048
#define BATCH    2
#define ROWS     (BATCH * SEQ)   // 4096

typedef __attribute__((ext_vector_type(16))) __bf16 v16bf;
typedef __attribute__((ext_vector_type(8)))  float  v8f;
typedef __attribute__((ext_vector_type(2)))  __bf16 bf16x2;
typedef __attribute__((ext_vector_type(4)))  __bf16 bf16x4;

union Frag { v16bf v; unsigned short u[16]; };

// native packed f32 -> bf16 conversions (lower to v_cvt_pk_bf16_f32)
__device__ __forceinline__ unsigned short f2bf(float f) {
  union { __bf16 b; unsigned short u; } cv;
  cv.b = (__bf16)f;
  return cv.u;
}
__device__ __forceinline__ unsigned f2bf2(float a, float b) {
  union { bf16x2 v; unsigned u; } cv;
  cv.v.x = (__bf16)a; cv.v.y = (__bf16)b;
  return cv.u;
}
__device__ __forceinline__ uint2 f2bf4(float4 v) {
  union { bf16x4 b; uint2 u; } cv;
  cv.b.x = (__bf16)v.x; cv.b.y = (__bf16)v.y;
  cv.b.z = (__bf16)v.z; cv.b.w = (__bf16)v.w;
  return cv.u;
}
__device__ __forceinline__ unsigned pk2(unsigned short lo, unsigned short hi) {
  return (unsigned)lo | ((unsigned)hi << 16);
}

__device__ __forceinline__ v8f wmma_bf16(const Frag& a, const Frag& b, v8f c) {
  return __builtin_amdgcn_wmma_f32_16x16x32_bf16(
      false, a.v, false, b.v, (short)0, c, false, false);
}

// A-fragment: u[0..7] at p, u[8..15] at p+16 (ISA 16-bit A layout, K-interleave)
__device__ __forceinline__ void ldA_frag(Frag& f, const unsigned short* p) {
  *(uint4*)&f.u[0] = *(const uint4*)p;
  *(uint4*)&f.u[8] = *(const uint4*)(p + 16);
}
// B-fragment: u[0..7] at p, u[8..15] at p+16 (K-interleave, matches A)
__device__ __forceinline__ void ldB_frag(Frag& f, const unsigned short* p) {
  *(uint4*)&f.u[0] = *(const uint4*)p;
  *(uint4*)&f.u[8] = *(const uint4*)(p + 16);
}

__device__ __forceinline__ unsigned lds_addr(const void* p) {
  return (unsigned)(unsigned long long)p;   // low 32 bits = LDS offset
}

// ---------------------------------------------------------------------------
// GEMM tiling: block 128x64, 8 waves (4m x 2n), wave tile 32x32, TK=64
// ---------------------------------------------------------------------------
#define TM 128
#define TN 64
#define TK 64
#define LDA  (TK + 8)   // 72
#define LDBT (TK + 8)   // 72, BT is [col][k]
#define LDC  (TN + 4)

// stage a 64x64 f32 W tile TRANSPOSED into lBT[col][k], k-pairs packed b32
__device__ __forceinline__ void stage_wt(unsigned short* lBT,
                                         const float* __restrict__ W,
                                         int k0, int col0, int tid) {
#pragma unroll
  for (int it = 0; it < 2; it++) {
    int g  = tid + it * 256;             // 512 groups: 2 k-rows x 4 cols
    int rp = g >> 4, c4 = (g & 15) * 4;  // k-row pair, col quad
    int r0 = rp * 2;
    const float* w0 = &W[(size_t)(k0 + r0) * D_MODEL + col0 + c4];
    float4 v0 = *(const float4*)w0;
    float4 v1 = *(const float4*)(w0 + D_MODEL);
    *(unsigned*)&lBT[(c4 + 0) * LDBT + r0] = f2bf2(v0.x, v1.x);
    *(unsigned*)&lBT[(c4 + 1) * LDBT + r0] = f2bf2(v0.y, v1.y);
    *(unsigned*)&lBT[(c4 + 2) * LDBT + r0] = f2bf2(v0.z, v1.z);
    *(unsigned*)&lBT[(c4 + 3) * LDBT + r0] = f2bf2(v0.w, v1.w);
  }
}

// ---------------------------------------------------------------------------
// Kernel 1: Y = X @ W + b  (f32 in, bf16 WMMA), optional RoPE, bf16 out
// ---------------------------------------------------------------------------
__global__ __launch_bounds__(256)
void qkv_rope_kernel(const float* __restrict__ X,
                     const float* __restrict__ W,
                     const float* __restrict__ bias,
                     unsigned short* __restrict__ Out,   // [B,H,S,hd] bf16
                     int do_rope)
{
  __shared__ unsigned short lA[TM * LDA];
  __shared__ unsigned short lBT[TN * LDBT];
  __shared__ float          lC[TM * LDC];

  const int tid  = threadIdx.x;
  const int lane = tid & 31;
  const int wid  = tid >> 5;
  const int l    = lane & 15;
  const int hi   = lane >> 4;

  const int row0 = blockIdx.x * TM;
  const int head = blockIdx.y;
  const int col0 = head * TN;

  const int mr = (wid & 3) * 32;
  const int nr = (wid >> 2) * 32;

  v8f acc[2][2] = {};

  for (int k0 = 0; k0 < D_MODEL; k0 += TK) {
    // stage X tile 128x64: float4 -> packed bf16x4 (uint2) stores
    for (int q = tid; q < (TM * TK) / 4; q += 256) {
      int r = q >> 4, c4 = (q & 15) * 4;
      float4 v = *(const float4*)&X[(size_t)(row0 + r) * D_MODEL + k0 + c4];
      *(uint2*)&lA[r * LDA + c4] = f2bf4(v);
    }
    stage_wt(lBT, W, k0, col0, tid);
    if (k0 + TK < D_MODEL)
      __builtin_prefetch(&X[(size_t)(row0 + (tid >> 1)) * D_MODEL + k0 + TK], 0, 0);
    __syncthreads();

    Frag a[2][2], b[2][2];
#pragma unroll
    for (int mi = 0; mi < 2; mi++)
#pragma unroll
      for (int kc = 0; kc < 2; kc++)
        ldA_frag(a[mi][kc], &lA[(mr + mi * 16 + l) * LDA + kc * 32 + hi * 8]);
#pragma unroll
    for (int ni = 0; ni < 2; ni++)
#pragma unroll
      for (int kc = 0; kc < 2; kc++)
        ldB_frag(b[ni][kc], &lBT[(nr + ni * 16 + l) * LDBT + kc * 32 + hi * 8]);
#pragma unroll
    for (int mi = 0; mi < 2; mi++)
#pragma unroll
      for (int ni = 0; ni < 2; ni++) {
        acc[mi][ni] = wmma_bf16(a[mi][0], b[ni][0], acc[mi][ni]);
        acc[mi][ni] = wmma_bf16(a[mi][1], b[ni][1], acc[mi][ni]);
      }
    __syncthreads();
  }

  // accumulators + bias -> LDS f32 staging (RoPE pairs j and j+32 in-tile)
#pragma unroll
  for (int mi = 0; mi < 2; mi++)
#pragma unroll
    for (int ni = 0; ni < 2; ni++)
#pragma unroll
      for (int i = 0; i < 8; i++) {
        int r = mr + mi * 16 + 8 * hi + i;
        int c = nr + ni * 16 + l;
        lC[r * LDC + c] = acc[mi][ni][i] + bias[col0 + c];
      }
  __syncthreads();

  const int bidx = row0 >> 11;
  const int s0   = row0 & (SEQ - 1);
  unsigned short* outBase =
      Out + ((size_t)bidx * N_HEADS + head) * SEQ * HEAD_DIM;

  if (do_rope) {
    // process rotary pairs two-at-a-time -> packed b32 stores
    for (int p = tid; p < TM * 16; p += 256) {
      int r = p >> 4, j2 = (p & 15) * 2;      // j2 in {0,2,...,30}
      int s = s0 + r;
      float or_[2], oi_[2];
#pragma unroll
      for (int e = 0; e < 2; e++) {
        int j = j2 + e;
        float inv = __expf(-(float)(2 * j) * (9.210340371976184f / 64.0f));
        float sn, cs;
        __sincosf((float)s * inv, &sn, &cs);
        float xr = lC[r * LDC + j];
        float xi = lC[r * LDC + 32 + j];
        or_[e] = xr * cs - xi * sn;
        oi_[e] = xi * cs + xr * sn;
      }
      unsigned short* o = outBase + (size_t)s * HEAD_DIM;
      *(unsigned*)&o[j2]      = f2bf2(or_[0], or_[1]);
      *(unsigned*)&o[32 + j2] = f2bf2(oi_[0], oi_[1]);
    }
  } else {
    for (int p = tid; p < TM * 16; p += 256) {
      int r = p >> 4, j4 = (p & 15) * 4;
      float4 v = { lC[r * LDC + j4],     lC[r * LDC + j4 + 1],
                   lC[r * LDC + j4 + 2], lC[r * LDC + j4 + 3] };
      *(uint2*)&outBase[(size_t)(s0 + r) * HEAD_DIM + j4] = f2bf4(v);
    }
  }
}

// ---------------------------------------------------------------------------
// Kernel 2: flash attention, grid (S/128, B*H), 8 waves; 64-key KV tiles.
// K tile staged via GLOBAL_LOAD_ASYNC_TO_LDS_B128 (ASYNCcnt), V transposed
// with key-pair b32 packing, P bounced through per-wave LDS for re-swizzle.
// ---------------------------------------------------------------------------
#define KVT   64
#define LVT   (KVT + 8)   // 72, Vt is [d][key]
#define LPP   (KVT + 8)   // 72, P  is [qrow][key]

__global__ __launch_bounds__(256)
void flash_attn_kernel(const unsigned short* __restrict__ Q,
                       const unsigned short* __restrict__ K,
                       const unsigned short* __restrict__ V,
                       unsigned short* __restrict__ O)    // [B,S,D] bf16
{
  __shared__ unsigned short lK[KVT * HEAD_DIM];     // [key][d], 8 KB
  __shared__ unsigned short lVt[HEAD_DIM * LVT];    // [d][key]
  __shared__ unsigned short lP[8 * 16 * LPP];

  const int tid  = threadIdx.x;
  const int lane = tid & 31;
  const int wid  = tid >> 5;
  const int l    = lane & 15;
  const int hi   = lane >> 4;

  const int bh = blockIdx.y;
  const int b_ = bh >> 4, h = bh & 15;
  const size_t base = (size_t)bh * SEQ * HEAD_DIM;
  const unsigned short* Qp = Q + base;
  const unsigned short* Kp = K + base;
  const unsigned short* Vp = V + base;

  const int qr = blockIdx.x * 128 + wid * 16;

  // Q fragments (two K=32 chunks) via contiguous 16B global loads
  Frag aq[2];
  {
    const uint4* qp = (const uint4*)(Qp + (size_t)(qr + l) * HEAD_DIM);
    *(uint4*)&aq[0].u[0] = qp[hi];
    *(uint4*)&aq[0].u[8] = qp[2 + hi];
    *(uint4*)&aq[1].u[0] = qp[4 + hi];
    *(uint4*)&aq[1].u[8] = qp[6 + hi];
  }

  v8f   o[4] = {};
  float m[8], lsum[8];
#pragma unroll
  for (int i = 0; i < 8; i++) { m[i] = -1e30f; lsum[i] = 0.0f; }

  const float scale = 0.125f;
  const unsigned lK0 = lds_addr(lK) + (unsigned)tid * 16u;

  for (int kb = 0; kb < SEQ; kb += KVT) {
    // K tile (8 KB, pure copy): async DMA global -> LDS, 2 x b128 per thread
    {
      const unsigned short* gk = Kp + (size_t)kb * HEAD_DIM;
      const unsigned voff = (unsigned)tid * 16u;
      asm volatile(
          "global_load_async_to_lds_b128 %0, %1, %2 offset:0\n\t"
          "global_load_async_to_lds_b128 %0, %1, %2 offset:4096"
          :: "v"(lK0), "v"(voff), "s"(gk) : "memory");
    }
    // V tile transposed: key-pair packing -> b32 LDS stores
    {
      const uint4* gv = (const uint4*)(Vp + (size_t)kb * HEAD_DIM);
      int kp2 = tid >> 3, d0 = (tid & 7) * 8;     // key pair, d octet
      union { uint4 q; unsigned short us[8]; } t0, t1;
      t0.q = gv[(kp2 * 2 + 0) * 8 + (tid & 7)];
      t1.q = gv[(kp2 * 2 + 1) * 8 + (tid & 7)];
#pragma unroll
      for (int e = 0; e < 8; e++)
        *(unsigned*)&lVt[(d0 + e) * LVT + kp2 * 2] = pk2(t0.us[e], t1.us[e]);
    }
    if (kb + KVT < SEQ)
      __builtin_prefetch(Kp + (size_t)(kb + KVT) * HEAD_DIM + tid * 8, 0, 0);
    asm volatile("s_wait_asynccnt 0" ::: "memory");
    __syncthreads();

    // scores: 4 key-tiles x (2 d-chunks) = 8 WMMA
    v8f sc[4];
#pragma unroll
    for (int t = 0; t < 4; t++) {
      Frag bk0, bk1;
      const unsigned short* kp = &lK[(t * 16 + l) * HEAD_DIM];
      ldB_frag(bk0, kp + hi * 8);          // d in [0,32)
      ldB_frag(bk1, kp + 32 + hi * 8);     // d in [32,64)
      v8f c = {};
      c = wmma_bf16(aq[0], bk0, c);
      c = wmma_bf16(aq[1], bk1, c);
      sc[t] = c;
    }

    // online softmax; row r = 8*hi + i lives in one 16-lane half
    unsigned short* pw = lP + wid * (16 * LPP);
#pragma unroll
    for (int i = 0; i < 8; i++) {
      float s0 = sc[0][i] * scale, s1 = sc[1][i] * scale;
      float s2 = sc[2][i] * scale, s3 = sc[3][i] * scale;
      float mx = fmaxf(fmaxf(s0, s1), fmaxf(s2, s3));
#pragma unroll
      for (int msk = 1; msk < 16; msk <<= 1)
        mx = fmaxf(mx, __shfl_xor(mx, msk, 32));
      float mn = fmaxf(m[i], mx);
      float alpha = __expf(m[i] - mn);
      m[i] = mn;
      float p0 = __expf(s0 - mn), p1 = __expf(s1 - mn);
      float p2 = __expf(s2 - mn), p3 = __expf(s3 - mn);
      float rs = (p0 + p1) + (p2 + p3);
#pragma unroll
      for (int msk = 1; msk < 16; msk <<= 1)
        rs += __shfl_xor(rs, msk, 32);
      lsum[i] = lsum[i] * alpha + rs;
#pragma unroll
      for (int t = 0; t < 4; t++) o[t][i] *= alpha;
      int r = (8 * hi + i) * LPP + l;
      pw[r]      = f2bf(p0);
      pw[r + 16] = f2bf(p1);
      pw[r + 32] = f2bf(p2);
      pw[r + 48] = f2bf(p3);
    }
    // per-wave LDS RAW fence (LDS in-order within a wave)
    asm volatile("s_wait_dscnt 0" ::: "memory");

    Frag aP[2];
#pragma unroll
    for (int kc = 0; kc < 2; kc++)
      ldA_frag(aP[kc], &pw[l * LPP + kc * 32 + hi * 8]);

    // O += P @ V : 4 d-tiles x 2 key-chunks = 8 WMMA
#pragma unroll
    for (int t = 0; t < 4; t++) {
      Frag bv0, bv1;
      const unsigned short* vp = &lVt[(t * 16 + l) * LVT];
      ldB_frag(bv0, vp + hi * 8);
      ldB_frag(bv1, vp + 32 + hi * 8);
      o[t] = wmma_bf16(aP[0], bv0, o[t]);
      o[t] = wmma_bf16(aP[1], bv1, o[t]);
    }
    __syncthreads();
  }

  // normalize, store bf16 in [B,S,D] layout for the output projection
#pragma unroll
  for (int i = 0; i < 8; i++) {
    float inv = 1.0f / lsum[i];
    int q = qr + 8 * hi + i;
    size_t rowoff = ((size_t)b_ * SEQ + q) * D_MODEL + h * HEAD_DIM;
#pragma unroll
    for (int t = 0; t < 4; t++)
      O[rowoff + t * 16 + l] = f2bf(o[t][i] * inv);
  }
}

// ---------------------------------------------------------------------------
// Kernel 3: Out = A @ Wo + bo   (A bf16 [4096,1024], Out f32)
// A tile staged with GLOBAL_LOAD_ASYNC_TO_LDS_B128.
// ---------------------------------------------------------------------------
__global__ __launch_bounds__(256)
void out_proj_kernel(const unsigned short* __restrict__ A,
                     const float* __restrict__ W,
                     const float* __restrict__ bias,
                     float* __restrict__ Out)
{
  __shared__ unsigned short lA[TM * LDA];
  __shared__ unsigned short lBT[TN * LDBT];

  const int tid  = threadIdx.x;
  const int lane = tid & 31;
  const int wid  = tid >> 5;
  const int l    = lane & 15;
  const int hi   = lane >> 4;

  const int row0 = blockIdx.x * TM;
  const int col0 = blockIdx.y * TN;
  const int mr = (wid & 3) * 32;
  const int nr = (wid >> 2) * 32;

  v8f acc[2][2] = {};
  const unsigned lAbase = lds_addr(lA);

  for (int k0 = 0; k0 < D_MODEL; k0 += TK) {
    // stage A tile (bf16 copy) via async DMA: 4 x b128 chunks per thread
    {
      const unsigned short* gA = A + (size_t)row0 * D_MODEL + k0;
#pragma unroll
      for (int it = 0; it < 4; it++) {
        int q = tid + it * 256;            // 1024 chunks of 16B
        int r = q >> 3, c8 = (q & 7) * 8;  // row, element offset
        unsigned voff = (unsigned)((r * D_MODEL + c8) * 2);
        unsigned dst  = lAbase + (unsigned)((r * LDA + c8) * 2);
        asm volatile("global_load_async_to_lds_b128 %0, %1, %2 offset:0"
                     :: "v"(dst), "v"(voff), "s"(gA) : "memory");
      }
    }
    stage_wt(lBT, W, k0, col0, tid);
    asm volatile("s_wait_asynccnt 0" ::: "memory");
    __syncthreads();

    Frag a[2][2], b[2][2];
#pragma unroll
    for (int mi = 0; mi < 2; mi++)
#pragma unroll
      for (int kc = 0; kc < 2; kc++)
        ldA_frag(a[mi][kc], &lA[(mr + mi * 16 + l) * LDA + kc * 32 + hi * 8]);
#pragma unroll
    for (int ni = 0; ni < 2; ni++)
#pragma unroll
      for (int kc = 0; kc < 2; kc++)
        ldB_frag(b[ni][kc], &lBT[(nr + ni * 16 + l) * LDBT + kc * 32 + hi * 8]);
#pragma unroll
    for (int mi = 0; mi < 2; mi++)
#pragma unroll
      for (int ni = 0; ni < 2; ni++) {
        acc[mi][ni] = wmma_bf16(a[mi][0], b[ni][0], acc[mi][ni]);
        acc[mi][ni] = wmma_bf16(a[mi][1], b[ni][1], acc[mi][ni]);
      }
    __syncthreads();
  }

#pragma unroll
  for (int mi = 0; mi < 2; mi++)
#pragma unroll
    for (int ni = 0; ni < 2; ni++)
#pragma unroll
      for (int i = 0; i < 8; i++) {
        int r = row0 + mr + mi * 16 + 8 * hi + i;
        int c = col0 + nr + ni * 16 + l;
        Out[(size_t)r * D_MODEL + c] = acc[mi][ni][i] + bias[c];
      }
}

// ---------------------------------------------------------------------------
extern "C" void kernel_launch(void* const* d_in, const int* in_sizes, int n_in,
                              void* d_out, int out_size, void* d_ws, size_t ws_size,
                              hipStream_t stream) {
  const float* x  = (const float*)d_in[0];
  const float* Wq = (const float*)d_in[1];
  const float* bq = (const float*)d_in[2];
  const float* Wk = (const float*)d_in[3];
  const float* bk = (const float*)d_in[4];
  const float* Wv = (const float*)d_in[5];
  const float* bv = (const float*)d_in[6];
  const float* Wo = (const float*)d_in[7];
  const float* bo = (const float*)d_in[8];
  float* out = (float*)d_out;

  unsigned short* ws = (unsigned short*)d_ws;
  const size_t QN = (size_t)BATCH * N_HEADS * SEQ * HEAD_DIM;  // 4,194,304
  unsigned short* Qb = ws;
  unsigned short* Kb = ws + QN;
  unsigned short* Vb = ws + 2 * QN;
  unsigned short* Ab = ws + 3 * QN;

  dim3 gGemm(ROWS / TM, D_MODEL / TN);       // (32, 16)
  qkv_rope_kernel<<<gGemm, 256, 0, stream>>>(x, Wq, bq, Qb, 1);
  qkv_rope_kernel<<<gGemm, 256, 0, stream>>>(x, Wk, bk, Kb, 1);
  qkv_rope_kernel<<<gGemm, 256, 0, stream>>>(x, Wv, bv, Vb, 0);

  dim3 gAttn(SEQ / 128, BATCH * N_HEADS);    // (16, 32)
  flash_attn_kernel<<<gAttn, 256, 0, stream>>>(Qb, Kb, Vb, Ab);

  out_proj_kernel<<<gGemm, 256, 0, stream>>>(Ab, Wo, bo, out);
}
